// PCRecoveryModule_30064771072406
// MI455X (gfx1250) — matrix operand order repaired and checked by reading, not compiled, not run on hardware
//
#include <hip/hip_runtime.h>
#include <hip/hip_bf16.h>

typedef __bf16 bf16_t;
typedef bf16_t v8bf  __attribute__((ext_vector_type(8)));
typedef bf16_t v16bf __attribute__((ext_vector_type(16)));
typedef float  v8f   __attribute__((ext_vector_type(8)));

// ---------------------------------------------------------------- utilities

__global__ void cvt_f32_bf16(const float* __restrict__ x, bf16_t* __restrict__ y, size_t n) {
  size_t i = (size_t)blockIdx.x * blockDim.x + threadIdx.x;
  size_t stride = (size_t)gridDim.x * blockDim.x;
  for (; i < n; i += stride) y[i] = (bf16_t)x[i];
}

// WT[b][n][k] = W[b][k][n]  (batch of K x N matrices -> N x K bf16)
__global__ void transpose_f32_bf16(const float* __restrict__ W, bf16_t* __restrict__ WT,
                                   int K, int N, size_t total) {
  size_t e = (size_t)blockIdx.x * blockDim.x + threadIdx.x;
  size_t stride = (size_t)gridDim.x * blockDim.x;
  size_t kn = (size_t)K * N;
  for (; e < total; e += stride) {
    size_t b = e / kn;
    size_t r = e - b * kn;
    int k = (int)(r / (size_t)N);
    int n = (int)(r - (size_t)k * N);
    WT[b * kn + (size_t)n * K + k] = (bf16_t)W[e];
  }
}

// BN (eval) + ReLU:  writes f32 in place (residual path) and bf16 copy (GEMM A path)
__global__ void bn_relu(float* __restrict__ S, bf16_t* __restrict__ Gbf,
                        const float* __restrict__ g, const float* __restrict__ b,
                        const float* __restrict__ m, const float* __restrict__ v,
                        size_t total, int cmask) {
  size_t i = (size_t)blockIdx.x * blockDim.x + threadIdx.x;
  size_t stride = (size_t)gridDim.x * blockDim.x;
  for (; i < total; i += stride) {
    int c = (int)(i & (size_t)cmask);
    float s = g[c] * rsqrtf(v[c] + 1e-5f);
    float x = (S[i] - m[c]) * s + b[c];
    x = fmaxf(x, 0.0f);
    S[i] = x;
    Gbf[i] = (bf16_t)x;
  }
}

// out[n,3] = F(bf16)[n,64] @ Wf[64,3]
__global__ void final_proj(const bf16_t* __restrict__ F, const float* __restrict__ Wf,
                           float* __restrict__ out, int n) {
  int r = blockIdx.x * blockDim.x + threadIdx.x;
  if (r >= n) return;
  const bf16_t* f = F + (size_t)r * 64;
  float o0 = 0.f, o1 = 0.f, o2 = 0.f;
#pragma unroll
  for (int k = 0; k < 64; ++k) {
    float x = (float)f[k];
    o0 += x * Wf[k * 3 + 0];
    o1 += x * Wf[k * 3 + 1];
    o2 += x * Wf[k * 3 + 2];
  }
  out[(size_t)r * 3 + 0] = o0;
  out[(size_t)r * 3 + 1] = o1;
  out[(size_t)r * 3 + 2] = o2;
}

// ---------------------------------------------------------------- WMMA GEMMs
// A: [M,K] bf16 row-major.  BT: [N,K] bf16 row-major (pre-transposed B).
// A fragment: lane lr holds row (mrow+lr); elems 0..15 -> K = kk + lh*16 + j
// (contiguous, matching B).
// B fragment: lane lr = col N; lanes 0-15 K=kk..kk+15, lanes 16-31 K=kk+16..kk+31
// -> one contiguous 32B load per lane from BT.

__device__ __forceinline__ v16bf load_a_frag(const bf16_t* Alo, int kk) {
  v8bf lo = *(const v8bf*)(Alo + kk);
  v8bf hi = *(const v8bf*)(Alo + kk + 8);
  v16bf a;
#pragma unroll
  for (int i = 0; i < 8; ++i) { a[i] = lo[i]; a[i + 8] = hi[i]; }
  return a;
}

#define NTILE 4  // 16x64 output tile per wave: A fragment reused 4x per k-step

// C(bf16) = A @ BT^T + bias
__global__ void gemm_bias(const bf16_t* __restrict__ A, const bf16_t* __restrict__ BT,
                          const float* __restrict__ bias, bf16_t* __restrict__ C,
                          int M, int N, int K) {
  const int lane = threadIdx.x;
  const int mrow = (blockIdx.x * blockDim.y + threadIdx.y) * 16;
  if (mrow >= M) return;                       // wave-uniform: EXEC stays all-1s
  const int nbase = blockIdx.y * (16 * NTILE);
  const int lr = lane & 15, lh = lane >> 4;
  int arow = mrow + lr; if (arow > M - 1) arow = M - 1;
  const bf16_t* Alo = A + (size_t)arow * K + lh * 16;
  const bf16_t* Bp[NTILE];
#pragma unroll
  for (int t = 0; t < NTILE; ++t)
    Bp[t] = BT + (size_t)(nbase + t * 16 + lr) * K + lh * 16;
  v8f acc[NTILE];
#pragma unroll
  for (int t = 0; t < NTILE; ++t)
#pragma unroll
    for (int i = 0; i < 8; ++i) acc[t][i] = 0.f;
  for (int kk = 0; kk < K; kk += 32) {
    v16bf a = load_a_frag(Alo, kk);
#pragma unroll
    for (int t = 0; t < NTILE; ++t) {
      v16bf b = *(const v16bf*)(Bp[t] + kk);
      acc[t] = __builtin_amdgcn_wmma_f32_16x16x32_bf16(false, a, false, b,
                                                       (short)0, acc[t], false, false);
    }
  }
#pragma unroll
  for (int t = 0; t < NTILE; ++t) {
    int col = nbase + t * 16 + lr;
    float bv = bias[col];
#pragma unroll
    for (int r = 0; r < 8; ++r) {
      int row = mrow + r + 8 * lh;
      if (row < M) C[(size_t)row * N + col] = (bf16_t)(acc[t][r] + bv);
    }
  }
}

// H(bf16) = G @ WsT^T + R(f32)   (1x1 conv + residual)
__global__ void gemm_res(const bf16_t* __restrict__ A, const bf16_t* __restrict__ BT,
                         const float* __restrict__ R, bf16_t* __restrict__ C,
                         int M, int N, int K) {
  const int lane = threadIdx.x;
  const int mrow = (blockIdx.x * blockDim.y + threadIdx.y) * 16;
  if (mrow >= M) return;
  const int nbase = blockIdx.y * (16 * NTILE);
  const int lr = lane & 15, lh = lane >> 4;
  int arow = mrow + lr; if (arow > M - 1) arow = M - 1;
  const bf16_t* Alo = A + (size_t)arow * K + lh * 16;
  const bf16_t* Bp[NTILE];
#pragma unroll
  for (int t = 0; t < NTILE; ++t)
    Bp[t] = BT + (size_t)(nbase + t * 16 + lr) * K + lh * 16;
  v8f acc[NTILE];
#pragma unroll
  for (int t = 0; t < NTILE; ++t)
#pragma unroll
    for (int i = 0; i < 8; ++i) acc[t][i] = 0.f;
  for (int kk = 0; kk < K; kk += 32) {
    v16bf a = load_a_frag(Alo, kk);
#pragma unroll
    for (int t = 0; t < NTILE; ++t) {
      v16bf b = *(const v16bf*)(Bp[t] + kk);
      acc[t] = __builtin_amdgcn_wmma_f32_16x16x32_bf16(false, a, false, b,
                                                       (short)0, acc[t], false, false);
    }
  }
#pragma unroll
  for (int t = 0; t < NTILE; ++t) {
    int col = nbase + t * 16 + lr;
#pragma unroll
    for (int r = 0; r < 8; ++r) {
      int row = mrow + r + 8 * lh;
      if (row < M)
        C[(size_t)row * N + col] = (bf16_t)(acc[t][r] + R[(size_t)row * N + col]);
    }
  }
}

// S(f32)[idx[k][row]] += (A @ WtT[k]^T)[row]   -- gather-GEMM-scatter, grid.z = kernel offset
__global__ void gemm_scatter(const bf16_t* __restrict__ A, const bf16_t* __restrict__ BTall,
                             const int* __restrict__ idxAll, float* __restrict__ S,
                             int Min, int N, int K) {
  const int koff = blockIdx.z;
  const bf16_t* BT = BTall + (size_t)koff * N * K;
  const int* idx = idxAll + (size_t)koff * Min;
  const int lane = threadIdx.x;
  const int mrow = (blockIdx.x * blockDim.y + threadIdx.y) * 16;
  if (mrow >= Min) return;
  const int nbase = blockIdx.y * (16 * NTILE);
  const int lr = lane & 15, lh = lane >> 4;
  int arow = mrow + lr; if (arow > Min - 1) arow = Min - 1;
  const bf16_t* Alo = A + (size_t)arow * K + lh * 16;
  const bf16_t* Bp[NTILE];
#pragma unroll
  for (int t = 0; t < NTILE; ++t)
    Bp[t] = BT + (size_t)(nbase + t * 16 + lr) * K + lh * 16;
  v8f acc[NTILE];
#pragma unroll
  for (int t = 0; t < NTILE; ++t)
#pragma unroll
    for (int i = 0; i < 8; ++i) acc[t][i] = 0.f;
  for (int kk = 0; kk < K; kk += 32) {
    v16bf a = load_a_frag(Alo, kk);
#pragma unroll
    for (int t = 0; t < NTILE; ++t) {
      v16bf b = *(const v16bf*)(Bp[t] + kk);
      acc[t] = __builtin_amdgcn_wmma_f32_16x16x32_bf16(false, a, false, b,
                                                       (short)0, acc[t], false, false);
    }
  }
#pragma unroll
  for (int r = 0; r < 8; ++r) {
    int row = mrow + r + 8 * lh;
    if (row < Min) {
      int o = idx[row];
#pragma unroll
      for (int t = 0; t < NTILE; ++t) {
        int col = nbase + t * 16 + lr;
        (void)__hip_atomic_fetch_add(&S[(size_t)o * N + col], acc[t][r],
                                     __ATOMIC_RELAXED, __HIP_MEMORY_SCOPE_AGENT);
      }
    }
  }
}

// ---------------------------------------------------------------- host side

static inline dim3 gemm_grid(int M, int N) {
  int mt = (M + 15) / 16;
  return dim3((unsigned)((mt + 3) / 4), (unsigned)(N / (16 * NTILE)), 1);
}

extern "C" void kernel_launch(void* const* d_in, const int* in_sizes, int n_in,
                              void* d_out, int out_size, void* d_ws, size_t ws_size,
                              hipStream_t stream) {
  // Input order from setup_inputs():
  const float* x_F  = (const float*)d_in[0];
  const float* fc_W = (const float*)d_in[1];
  const float* fc_b = (const float*)d_in[2];
  const int*   idx1 = (const int*)d_in[3];
  const int*   idx2 = (const int*)d_in[4];
  const int*   idx3 = (const int*)d_in[5];
  // d_in[6..8] = n1,n2,n3 device scalars (sizes recovered from in_sizes instead)
  const float* Wt1 = (const float*)d_in[9];
  const float* g1  = (const float*)d_in[10];
  const float* b1  = (const float*)d_in[11];
  const float* m1  = (const float*)d_in[12];
  const float* v1  = (const float*)d_in[13];
  const float* Ws1 = (const float*)d_in[14];
  const float* Wt2 = (const float*)d_in[15];
  const float* g2  = (const float*)d_in[16];
  const float* b2  = (const float*)d_in[17];
  const float* m2  = (const float*)d_in[18];
  const float* v2  = (const float*)d_in[19];
  const float* Ws2 = (const float*)d_in[20];
  const float* Wt3 = (const float*)d_in[21];
  const float* g3  = (const float*)d_in[22];
  const float* b3  = (const float*)d_in[23];
  const float* m3  = (const float*)d_in[24];
  const float* v3  = (const float*)d_in[25];
  const float* Ws3 = (const float*)d_in[26];
  const float* Wf  = (const float*)d_in[27];

  const int N0 = in_sizes[3] / 8;   // idx1 is [8, N0]
  const int n1 = in_sizes[4] / 8;   // idx2 is [8, n1]
  const int n2 = in_sizes[5] / 8;   // idx3 is [8, n2]
  const int n3 = out_size / 3;      // output is [n3, 3]

  // workspace carve-out
  size_t maxElems = (size_t)N0 * 256;
  if ((size_t)n1 * 256 > maxElems) maxElems = (size_t)n1 * 256;
  if ((size_t)n2 * 128 > maxElems) maxElems = (size_t)n2 * 128;
  if ((size_t)n3 * 64  > maxElems) maxElems = (size_t)n3 * 64;
  size_t maxS = (size_t)n1 * 256;
  if ((size_t)n2 * 128 > maxS) maxS = (size_t)n2 * 128;
  if ((size_t)n3 * 64  > maxS) maxS = (size_t)n3 * 64;

  char* w = (char*)d_ws;
  auto carve = [&](size_t bytes) { void* p = (void*)w; w += (bytes + 255) & ~(size_t)255; return p; };
  bf16_t* bfA   = (bf16_t*)carve(maxElems * 2);
  bf16_t* bfB   = (bf16_t*)carve(maxElems * 2);
  float*  S     = (float*)carve(maxS * 4);
  bf16_t* fcWbf = (bf16_t*)carve((size_t)256 * 256 * 2);
  bf16_t* Wt1T  = (bf16_t*)carve((size_t)8 * 256 * 256 * 2);
  bf16_t* Ws1T  = (bf16_t*)carve((size_t)256 * 256 * 2);
  bf16_t* Wt2T  = (bf16_t*)carve((size_t)8 * 256 * 128 * 2);
  bf16_t* Ws2T  = (bf16_t*)carve((size_t)128 * 128 * 2);
  bf16_t* Wt3T  = (bf16_t*)carve((size_t)8 * 128 * 64 * 2);
  bf16_t* Ws3T  = (bf16_t*)carve((size_t)64 * 64 * 2);

  const dim3 tb(256), wb(32, 4);
  auto gsz = [](size_t n) { size_t g = (n + 255) / 256; return dim3((unsigned)(g > 4096 ? 4096 : g)); };

  // --- weight prep (bf16 / transposed-to-[N,K]) ---
  cvt_f32_bf16<<<gsz((size_t)N0 * 256), tb, 0, stream>>>(x_F, bfA, (size_t)N0 * 256);
  cvt_f32_bf16<<<gsz(256 * 256), tb, 0, stream>>>(fc_W, fcWbf, (size_t)256 * 256); // already [N,K]
  transpose_f32_bf16<<<gsz(8 * 256 * 256), tb, 0, stream>>>(Wt1, Wt1T, 256, 256, (size_t)8 * 256 * 256);
  transpose_f32_bf16<<<gsz(256 * 256), tb, 0, stream>>>(Ws1, Ws1T, 256, 256, (size_t)256 * 256);
  transpose_f32_bf16<<<gsz(8 * 256 * 128), tb, 0, stream>>>(Wt2, Wt2T, 256, 128, (size_t)8 * 256 * 128);
  transpose_f32_bf16<<<gsz(128 * 128), tb, 0, stream>>>(Ws2, Ws2T, 128, 128, (size_t)128 * 128);
  transpose_f32_bf16<<<gsz(8 * 128 * 64), tb, 0, stream>>>(Wt3, Wt3T, 128, 64, (size_t)8 * 128 * 64);
  transpose_f32_bf16<<<gsz(64 * 64), tb, 0, stream>>>(Ws3, Ws3T, 64, 64, (size_t)64 * 64);

  // --- fc: F0 = x @ fc_W.T + fc_b ---
  gemm_bias<<<gemm_grid(N0, 256), wb, 0, stream>>>(bfA, fcWbf, fc_b, bfB, N0, 256, 256);

  // --- block 1 ---
  hipMemsetAsync(S, 0, (size_t)n1 * 256 * 4, stream);
  { dim3 g = gemm_grid(N0, 256); g.z = 8;
    gemm_scatter<<<g, wb, 0, stream>>>(bfB, Wt1T, idx1, S, N0, 256, 256); }
  bn_relu<<<gsz((size_t)n1 * 256), tb, 0, stream>>>(S, bfA, g1, b1, m1, v1, (size_t)n1 * 256, 255);
  gemm_res<<<gemm_grid(n1, 256), wb, 0, stream>>>(bfA, Ws1T, S, bfB, n1, 256, 256);

  // --- block 2 ---
  hipMemsetAsync(S, 0, (size_t)n2 * 128 * 4, stream);
  { dim3 g = gemm_grid(n1, 128); g.z = 8;
    gemm_scatter<<<g, wb, 0, stream>>>(bfB, Wt2T, idx2, S, n1, 128, 256); }
  bn_relu<<<gsz((size_t)n2 * 128), tb, 0, stream>>>(S, bfA, g2, b2, m2, v2, (size_t)n2 * 128, 127);
  gemm_res<<<gemm_grid(n2, 128), wb, 0, stream>>>(bfA, Ws2T, S, bfB, n2, 128, 128);

  // --- block 3 ---
  hipMemsetAsync(S, 0, (size_t)n3 * 64 * 4, stream);
  { dim3 g = gemm_grid(n2, 64); g.z = 8;
    gemm_scatter<<<g, wb, 0, stream>>>(bfB, Wt3T, idx3, S, n2, 64, 128); }
  bn_relu<<<gsz((size_t)n3 * 64), tb, 0, stream>>>(S, bfA, g3, b3, m3, v3, (size_t)n3 * 64, 63);
  gemm_res<<<gemm_grid(n3, 64), wb, 0, stream>>>(bfA, Ws3T, S, bfB, n3, 64, 64);

  // --- final projection ---
  final_proj<<<dim3((n3 + 255) / 256), tb, 0, stream>>>(bfB, Wf, (float*)d_out, n3);
  (void)n_in; (void)ws_size;
}
